// SO3Layer_36163624632816
// MI455X (gfx1250) — matrix operand order repaired and checked by reading, not compiled, not run on hardware
//
#include <hip/hip_runtime.h>
#include <hip/hip_bf16.h>

typedef __attribute__((ext_vector_type(16))) _Float16 v16h;
typedef __attribute__((ext_vector_type(8)))  float    v8f;
typedef __attribute__((ext_vector_type(4)))  float    v4f;
typedef __attribute__((ext_vector_type(2)))  float    v2f;

#define IN_F    16
#define OUT_F   16
#define N_HARM  9
#define N_RBF   16
#define CUT_R   5.0f
#define WPB     4     // waves per block
#define EPW     16    // edges per wave (WMMA M)

// ---------------------------------------------------------------------------
// wave-level LDS fence: all per-wave LDS traffic, no block barrier needed
// ---------------------------------------------------------------------------
__device__ __forceinline__ void wave_lds_sync() {
    asm volatile("s_wait_dscnt 0" ::: "memory");
    __builtin_amdgcn_wave_barrier();
}

// fast silu: one v_exp_f32 + one v_rcp_f32 (no precise-div expansion)
__device__ __forceinline__ float silu_f(float x) {
    return x * __builtin_amdgcn_rcpf(1.0f + __expf(-x));
}

// ---------------------------------------------------------------------------
// Repack MLP weights into f16 WMMA B-fragments (ISA layout):
//   element e (0..15) of lane l  ->  K = (e&7) + 8*(l>=16) + 16*(e>=8),  N = l%16
// frag storage: frag * 512 halfs, lane-major (lane*16 + e)
// frags 0..3   : w1  (ntile 0..3, K real = 16, padded to 32 with zeros)
// frags 4..11  : w2  (f-4 = kt*4  + nt)
// frags 12..43 : w3  (f-12 = kt*16 + nt)
// ---------------------------------------------------------------------------
__global__ void pack_weights_kernel(const float* __restrict__ w1,
                                    const float* __restrict__ w2,
                                    const float* __restrict__ w3,
                                    _Float16* __restrict__ frags) {
    for (int idx = threadIdx.x + blockIdx.x * blockDim.x; idx < 44 * 512;
         idx += blockDim.x * gridDim.x) {
        int frag  = idx >> 9;
        int r     = idx & 511;
        int lane  = r >> 4;
        int e     = r & 15;
        int n_loc = lane & 15;
        int k_loc = (e & 7) + ((lane >> 4) << 3) + ((e >> 3) << 4);   // 0..31
        float v = 0.0f;
        if (frag < 4) {
            int n = frag * 16 + n_loc;
            if (k_loc < N_RBF) v = w1[n * N_RBF + k_loc];
        } else if (frag < 12) {
            int f = frag - 4;  int kt = f >> 2;  int nt = f & 3;
            v = w2[(nt * 16 + n_loc) * 64 + kt * 32 + k_loc];
        } else {
            int f = frag - 12; int kt = f >> 4;  int nt = f & 15;
            v = w3[(nt * 16 + n_loc) * 64 + kt * 32 + k_loc];
        }
        frags[idx] = (_Float16)v;
    }
}

__global__ void zero_kernel(v4f* __restrict__ p, int n4) {
    for (int i = threadIdx.x + blockIdx.x * blockDim.x; i < n4;
         i += blockDim.x * gridDim.x)
        p[i] = (v4f){0.0f, 0.0f, 0.0f, 0.0f};
}

// ---------------------------------------------------------------------------
// Edge kernel: 1 wave = 16 edges.  MLP on WMMA f16 (f32 accum, literal-0 C,
// bias in epilogue), einsum on packed-f32 VALU from per-wave LDS tiles,
// scatter via global_atomic_add_f32.
// ---------------------------------------------------------------------------
__global__ __launch_bounds__(128) void edge_kernel(
    const float* __restrict__ x,
    const int*   __restrict__ edge_index,
    const float* __restrict__ edge_vec,
    const float* __restrict__ b1,
    const float* __restrict__ b2,
    const float* __restrict__ b3,
    const float* __restrict__ rbf_widths,
    const _Float16* __restrict__ frags,
    float* __restrict__ out,
    int E)
{
    // x[src] tile, transposed to [edge][h][i] so the einsum reads 16
    // contiguous floats per (edge,h); sh is factored out of the i-sum.
    __shared__ float    s_xT[WPB][EPW * N_HARM * IN_F];   // 36 KB
    __shared__ _Float16 s_h[WPB][EPW * 64];               //  8 KB activations
    __shared__ float    s_rw[WPB][EPW * 16];              //  4 KB one rw tile
    __shared__ float    s_sh9[WPB][EPW][N_HARM];
    __shared__ float    s_cut[WPB][EPW];
    __shared__ int      s_src[WPB][EPW];
    __shared__ int      s_dst[WPB][EPW];

    const int lane  = threadIdx.x & 31;
    const int w     = threadIdx.x >> 5;
    const int tile  = blockIdx.x * WPB + w;
    const int eBase = tile * EPW;
    if (eBase >= E) return;                    // uniform per wave

    const int lhi  = lane >> 4;                // 0 / 1 half-wave
    const int llo  = lane & 15;
    int e = eBase + llo;
    const bool valid = (e < E);
    if (!valid) e = E - 1;

    // ---- per-edge geometry (lanes l and l+16 duplicate edge llo) ----
    float vx = edge_vec[e * 3 + 0];
    float vy = edge_vec[e * 3 + 1];
    float vz = edge_vec[e * 3 + 2];
    float d    = sqrtf(vx * vx + vy * vy + vz * vz);
    float rinv = __builtin_amdgcn_rcpf(fmaxf(d, 1e-12f));
    float ux = vx * rinv, uy = vy * rinv, uz = vz * rinv;

    const float C0 = 0.28209479177387814f, C1 = 0.4886025119029199f,
                C2 = 1.0925484305920792f,  C20 = 0.31539156525252005f,
                C22 = 0.5462742152960396f;
    float sh[9];
    sh[0] = C0;           sh[1] = C1 * uy;     sh[2] = C1 * uz;
    sh[3] = C1 * ux;      sh[4] = C2 * ux * uy; sh[5] = C2 * uy * uz;
    sh[6] = C20 * (3.0f * uz * uz - 1.0f);
    sh[7] = C2 * ux * uz; sh[8] = C22 * (ux * ux - uy * uy);

    float cut = 0.0f;
    if (valid && d < CUT_R)
        cut = 0.5f * (1.0f + __cosf(3.14159265358979323846f * d * (1.0f / CUT_R)));

    if (lane < 16) {
#pragma unroll
        for (int h = 0; h < 9; ++h) s_sh9[w][llo][h] = sh[h];
        s_cut[w][llo] = cut;
        s_src[w][llo] = edge_index[e];
        s_dst[w][llo] = edge_index[E + e];
    }

    // ---- A1 fragment (16x32 f16, real K=16): elem j -> K = j + 8*lhi ----
    // each lane computes only its own 8 RBFs (no register-array indexing)
    v16h a1;
    {
        const float* wp = rbf_widths + lhi * 8;   // contiguous per lane
#pragma unroll
        for (int j = 0; j < 8; ++j) {
            float c  = (float)(lhi * 8 + j) * (CUT_R / 15.0f);
            float wj = wp[j];
            float t  = d - c;
            float inv2w2 = __builtin_amdgcn_rcpf(2.0f * wj * wj);
            a1[j]     = (_Float16)__expf(-(t * t) * inv2w2);
            a1[j + 8] = (_Float16)0.0f;          // K = 16..31 padding
        }
    }
    wave_lds_sync();

    // ---- stage x[src] transposed: s_xT[el][h][i] = x[src][i][h] ----
    for (int t = lane; t < EPW * 144; t += 32) {
        int el  = t / 144;
        int rem = t - el * 144;            // = ii*9 + hh (contiguous gather)
        int ii  = rem / 9;
        int hh  = rem - ii * 9;
        s_xT[w][el * 144 + hh * 16 + ii] = x[(long)s_src[w][el] * 144 + rem];
    }

    const v16h* fB = (const v16h*)frags;   // frag f, lane l -> fB[f*32 + l]
    const v8f  cz  = {};                   // literal-0 C operand

    // ---- layer 1: (16x16) x (16x64), K padded to 32; bias in epilogue ----
#pragma unroll
    for (int nt = 0; nt < 4; ++nt) {
        v16h  b  = fB[nt * 32 + lane];
        float bv = b1[nt * 16 + llo];
        v8f   dv = __builtin_amdgcn_wmma_f32_16x16x32_f16(
                       false, a1, false, b, (short)0, cz, false, false);
#pragma unroll
        for (int v = 0; v < 8; ++v) {
            int row = v + (lhi << 3);
            s_h[w][row * 64 + nt * 16 + llo] = (_Float16)silu_f(dv[v] + bv);
        }
    }
    wave_lds_sync();

    // ---- reload h1 as A fragments (D layout -> A layout via LDS) ----
    v16h a2[2];
#pragma unroll
    for (int kt = 0; kt < 2; ++kt) {
        const _Float16* hp = &s_h[w][llo * 64 + kt * 32 + (lhi << 3)];
#pragma unroll
        for (int j = 0; j < 8; ++j) {
            a2[kt][j]     = hp[j];        // K = kt*32 + 8*lhi + j
            a2[kt][j + 8] = hp[j + 16];   // K = kt*32 + 16 + 8*lhi + j
        }
    }
    wave_lds_sync();   // reads done before h2 overwrites buffer

    // ---- layer 2: (16x64) x (64x64) ----
#pragma unroll
    for (int nt = 0; nt < 4; ++nt) {
        float bv = b2[nt * 16 + llo];
        v8f dv = __builtin_amdgcn_wmma_f32_16x16x32_f16(
                 false, a2[0], false, fB[(4 + nt) * 32 + lane],     (short)0, cz, false, false);
        dv = __builtin_amdgcn_wmma_f32_16x16x32_f16(
                 false, a2[1], false, fB[(4 + 4 + nt) * 32 + lane], (short)0, dv, false, false);
#pragma unroll
        for (int v = 0; v < 8; ++v) {
            int row = v + (lhi << 3);
            s_h[w][row * 64 + nt * 16 + llo] = (_Float16)silu_f(dv[v] + bv);
        }
    }
    wave_lds_sync();

    v16h a3[2];
#pragma unroll
    for (int kt = 0; kt < 2; ++kt) {
        const _Float16* hp = &s_h[w][llo * 64 + kt * 32 + (lhi << 3)];
#pragma unroll
        for (int j = 0; j < 8; ++j) {
            a3[kt][j]     = hp[j];
            a3[kt][j + 8] = hp[j + 16];
        }
    }

    // ---- layer 3: (16x64) x (64x256) tile-by-tile; tile nt == output row o.
    //      msg[e][o][h] = sh[h] * sum_i rw[e,o,i] * xT[e,h,i]
#pragma unroll
    for (int nt = 0; nt < 16; ++nt) {
        float bv = b3[nt * 16 + llo];
        v8f dv = __builtin_amdgcn_wmma_f32_16x16x32_f16(
                 false, a3[0], false, fB[(12 + nt) * 32 + lane],      (short)0, cz, false, false);
        dv = __builtin_amdgcn_wmma_f32_16x16x32_f16(
                 false, a3[1], false, fB[(12 + 16 + nt) * 32 + lane], (short)0, dv, false, false);
#pragma unroll
        for (int v = 0; v < 8; ++v) {
            int row = v + (lhi << 3);                         // edge in tile
            s_rw[w][row * 16 + llo] = (dv[v] + bv) * s_cut[w][row];  // col = i
        }
        wave_lds_sync();

        for (int cI = lane; cI < EPW * N_HARM; cI += 32) {
            int el = cI / 9;
            int hh = cI - el * 9;
            const v2f* rp = (const v2f*)&s_rw[w][el * 16];
            const v2f* xp = (const v2f*)&s_xT[w][el * 144 + hh * 16];
            v2f acc2 = {0.0f, 0.0f};
#pragma unroll
            for (int q = 0; q < 8; ++q) acc2 += rp[q] * xp[q];
            float acc = (acc2.x + acc2.y) * s_sh9[w][el][hh];
            atomicAdd(&out[(long)s_dst[w][el] * 144 + nt * 9 + hh], acc);
        }
        wave_lds_sync();   // einsum reads done before next tile overwrites s_rw
    }
}

// ---------------------------------------------------------------------------
// Node kernel: self-interaction + bias + LayerNorm(l=0) + SiLU, in place.
// 128 threads = 8 nodes x 16 o-lanes (each node group = half a wave32).
// ---------------------------------------------------------------------------
__global__ __launch_bounds__(128) void node_kernel(
    const float* __restrict__ x,
    const float* __restrict__ w_self,   // (3,16,16)
    const float* __restrict__ b_self,   // (16)
    const float* __restrict__ gamma,
    const float* __restrict__ beta,
    float* __restrict__ out, int N)
{
    __shared__ float s_w[3 * 16 * 16];
    __shared__ float s_x[8][144];
    const int tid = threadIdx.x;
    for (int i = tid; i < 768; i += 128) s_w[i] = w_self[i];

    const int g = tid >> 4;
    const int o = tid & 15;
    const int n = blockIdx.x * 8 + g;
    const bool valid = (n < N);
    if (valid) {
#pragma unroll
        for (int m = 0; m < 9; ++m)
            s_x[g][o * 9 + m] = x[(long)n * 144 + o * 9 + m];  // row i = o
    }
    __syncthreads();
    if (!valid) return;

    float tot[9];
#pragma unroll
    for (int m = 0; m < 9; ++m) {
        int l = (m == 0) ? 0 : ((m < 4) ? 1 : 2);
        float acc = 0.0f;
#pragma unroll
        for (int i = 0; i < 16; ++i)
            acc += s_x[g][i * 9 + m] * s_w[l * 256 + o * 16 + i];
        tot[m] = out[(long)n * 144 + o * 9 + m] + acc;
    }
    tot[0] += b_self[o];

#pragma unroll
    for (int m = 1; m < 9; ++m)
        out[(long)n * 144 + o * 9 + m] = tot[m];

    // LayerNorm over the 16 o-lanes of this node (width-16 shuffles)
    float s0 = tot[0];
    float sum = s0;
#pragma unroll
    for (int mask = 1; mask < 16; mask <<= 1) sum += __shfl_xor(sum, mask, 16);
    float mu = sum * (1.0f / 16.0f);
    float df = s0 - mu;
    float v2 = df * df;
#pragma unroll
    for (int mask = 1; mask < 16; mask <<= 1) v2 += __shfl_xor(v2, mask, 16);
    float var = v2 * (1.0f / 16.0f);
    float sn  = df * rsqrtf(var + 1e-5f) * gamma[o] + beta[o];
    out[(long)n * 144 + o * 9 + 0] = silu_f(sn);
}

// ---------------------------------------------------------------------------
extern "C" void kernel_launch(void* const* d_in, const int* in_sizes, int n_in,
                              void* d_out, int out_size, void* d_ws, size_t ws_size,
                              hipStream_t stream) {
    const float* x    = (const float*)d_in[0];
    const int*   ei   = (const int*)  d_in[1];
    const float* ev   = (const float*)d_in[2];
    const float* w1   = (const float*)d_in[3];
    const float* b1   = (const float*)d_in[4];
    const float* w2   = (const float*)d_in[5];
    const float* b2   = (const float*)d_in[6];
    const float* w3   = (const float*)d_in[7];
    const float* b3   = (const float*)d_in[8];
    const float* wid  = (const float*)d_in[9];
    const float* wsf  = (const float*)d_in[10];
    const float* bsf  = (const float*)d_in[11];
    const float* gam  = (const float*)d_in[12];
    const float* bet  = (const float*)d_in[13];
    float*    out   = (float*)d_out;
    _Float16* frags = (_Float16*)d_ws;   // 44 frags * 512 halfs = 44 KB

    const int E = in_sizes[1] / 2;
    const int N = in_sizes[0] / (IN_F * N_HARM);

    pack_weights_kernel<<<11, 256, 0, stream>>>(w1, w2, w3, frags);
    zero_kernel<<<1024, 256, 0, stream>>>((v4f*)out, out_size / 4);

    const int tiles   = (E + EPW - 1) / EPW;
    const int eblocks = (tiles + WPB - 1) / WPB;
    edge_kernel<<<eblocks, 128, 0, stream>>>(x, ei, ev, b1, b2, b3, wid,
                                             frags, out, E);

    const int nblocks = (N + 7) / 8;
    node_kernel<<<nblocks, 128, 0, stream>>>(x, wsf, bsf, gam, bet, out, N);
}
